// PreRoPEQuantAttention_41858751267193
// MI455X (gfx1250) — compile-verified, matrix operand order checked by reading, not run on hardware
//
#include <hip/hip_runtime.h>
#include <math.h>
#include <float.h>

typedef __attribute__((ext_vector_type(16))) __bf16          v16bf;
typedef __attribute__((ext_vector_type(16))) unsigned short  v16u;
typedef __attribute__((ext_vector_type(8)))  float           v8f;

// ---------------- helpers ----------------

__device__ __forceinline__ unsigned short f2bf(float f) {
    __bf16 h = (__bf16)f;                         // fptrunc -> native cvt
    return __builtin_bit_cast(unsigned short, h);
}

__device__ __forceinline__ float bf2f(unsigned short u) {
    return __uint_as_float(((unsigned int)u) << 16);
}

// Fragment from pre-converted bf16 data (two 16B loads).
// ISA 16-bit layout: lane half h: elements e<8 -> K = h*8+e ; e>=8 -> K = 16+h*8+(e-8)
__device__ __forceinline__ v16bf frag_from_bf16(const unsigned short* p, int hf) {
    union { uint4 u4[2]; v16bf b; } x;
    x.u4[0] = *(const uint4*)(p + hf * 8);
    x.u4[1] = *(const uint4*)(p + 16 + hf * 8);
    return x.b;
}

// Fragment built from fp32 values (used only for LDS-resident P in attention).
__device__ __forceinline__ v16bf frag_from_f32(const float* p, int hf) {
    union { v16u u; v16bf b; } x;
#pragma unroll
    for (int e = 0; e < 8; ++e) x.u[e]     = f2bf(p[hf * 8 + e]);
#pragma unroll
    for (int e = 0; e < 8; ++e) x.u[8 + e] = f2bf(p[16 + hf * 8 + e]);
    return x.b;
}

__device__ __forceinline__ v8f wmma_bf16(v16bf a, v16bf b, v8f c) {
    return __builtin_amdgcn_wmma_f32_16x16x32_bf16(
        false, a, false, b, (short)0, c, false, false);
}

// ---------------- fp32 -> bf16 bulk convert (4 elems / thread) ----------------

__global__ void cvt_f32_bf16(const float* __restrict__ s,
                             unsigned short* __restrict__ d, long long n4) {
    long long i = (long long)blockIdx.x * blockDim.x + threadIdx.x;
    if (i >= n4) return;
    float4 f = ((const float4*)s)[i];
    uint2 o;
    o.x = ((unsigned)f2bf(f.y) << 16) | f2bf(f.x);
    o.y = ((unsigned)f2bf(f.w) << 16) | f2bf(f.z);
    ((uint2*)d)[i] = o;
}

// ---------------- GEMM: C[M,N] = A[M,K] * B[N,K]^T, bf16 operands ----------------
// One wave per 64x64 tile. grid = (N/64, M/64), block = 32.
// K-loop body: 16 x b128 load + 16 x v_wmma, no conversion VALU.

template <bool OUT_BF16>
__global__ __launch_bounds__(32) void gemm_bf16(const unsigned short* __restrict__ A,
                                                const unsigned short* __restrict__ B,
                                                void* __restrict__ Cout,
                                                int M, int N, int K) {
    const int lane = threadIdx.x;
    const int hf   = lane >> 4;
    const int lr   = lane & 15;
    const int tN   = blockIdx.x * 64;
    const int tM   = blockIdx.y * 64;

    const unsigned short* arow[4];
    const unsigned short* brow[4];
#pragma unroll
    for (int r = 0; r < 4; ++r) arow[r] = A + (size_t)(tM + r * 16 + lr) * K;
#pragma unroll
    for (int c = 0; c < 4; ++c) brow[c] = B + (size_t)(tN + c * 16 + lr) * K;

    v8f acc[4][4] = {};

    for (int k = 0; k < K; k += 32) {
        v16bf af[4], bf_[4];
#pragma unroll
        for (int r = 0; r < 4; ++r) {
            __builtin_prefetch(arow[r] + k + 512, 0, 0);
            af[r] = frag_from_bf16(arow[r] + k, hf);
        }
#pragma unroll
        for (int c = 0; c < 4; ++c) {
            __builtin_prefetch(brow[c] + k + 512, 0, 0);
            bf_[c] = frag_from_bf16(brow[c] + k, hf);
        }
#pragma unroll
        for (int r = 0; r < 4; ++r)
#pragma unroll
            for (int c = 0; c < 4; ++c)
                acc[r][c] = wmma_bf16(af[r], bf_[c], acc[r][c]);
    }

#pragma unroll
    for (int r = 0; r < 4; ++r)
#pragma unroll
        for (int c = 0; c < 4; ++c)
#pragma unroll
            for (int v = 0; v < 8; ++v) {
                int row = tM + r * 16 + v + hf * 8;
                int col = tN + c * 16 + lr;
                if (OUT_BF16)
                    ((unsigned short*)Cout)[(size_t)row * N + col] = f2bf(acc[r][c][v]);
                else
                    ((float*)Cout)[(size_t)row * N + col] = acc[r][c][v];
            }
}

// ---------------- fake_quant_sparse along one axis, in place ----------------
// One block per slice. element i of slice: data[blk*base_stride + i*elem_stride]

__global__ void fake_quant(float* __restrict__ data, int n,
                           long long base_stride, long long elem_stride) {
    __shared__ float srt[2048];
    __shared__ float rmax[256];
    __shared__ float rmin[256];

    float* base = data + (long long)blockIdx.x * base_stride;
    const int tid = threadIdx.x;

    for (int i = tid; i < n; i += 256) srt[i] = base[i * elem_stride];
    __syncthreads();

    for (int k = 2; k <= n; k <<= 1) {
        for (int j = k >> 1; j > 0; j >>= 1) {
            for (int i = tid; i < n; i += 256) {
                int ij = i ^ j;
                if (ij > i) {
                    float a = srt[i], b = srt[ij];
                    bool up = ((i & k) == 0);
                    if ((a > b) == up) { srt[i] = b; srt[ij] = a; }
                }
            }
            __syncthreads();
        }
    }

    const double thr = 0.9995;
    double ph = thr * (double)(n - 1);
    int ih = (int)ph; double fh = ph - (double)ih;
    int ih1 = (ih + 1 < n) ? ih + 1 : n - 1;
    float upper = (float)((double)srt[ih] + fh * ((double)srt[ih1] - (double)srt[ih]));

    double pl = (1.0 - thr) * (double)(n - 1);
    int il = (int)pl; double fl = pl - (double)il;
    int il1 = (il + 1 < n) ? il + 1 : n - 1;
    float lower = (float)((double)srt[il] + fl * ((double)srt[il1] - (double)srt[il]));

    float med = srt[(n - 1) >> 1];
    __syncthreads();

    float mx = -FLT_MAX, mn = FLT_MAX;
    for (int i = tid; i < n; i += 256) {
        float x = base[i * elem_stride];
        bool m = (x <= lower) || (x >= upper);
        float t = m ? med : x;
        mx = fmaxf(mx, t);
        mn = fminf(mn, t);
    }
    rmax[tid] = mx; rmin[tid] = mn;
    __syncthreads();
    for (int off = 128; off > 0; off >>= 1) {
        if (tid < off) {
            rmax[tid] = fmaxf(rmax[tid], rmax[tid + off]);
            rmin[tid] = fminf(rmin[tid], rmin[tid + off]);
        }
        __syncthreads();
    }
    float maxv = rmax[0], minv = rmin[0];
    float qx   = 15.0f / (maxv - minv);
    float offs = minv * qx;

    for (int i = tid; i < n; i += 256) {
        float x = base[i * elem_stride];
        bool m = (x <= lower) || (x >= upper);
        float inp = m ? 0.0f : x;
        float q = fminf(fmaxf(rintf(qx * inp - offs), 0.0f), 15.0f);
        float deq = (q + offs) / qx;
        float o = m ? x : deq;
        if (!isfinite(o)) o = 0.0f;   // nan_to_num(0,0,0)
        base[i * elem_stride] = o;
    }
}

// ---------------- RoPE ----------------

// in-place RoPE on a bf16 buffer (used for Q)
__global__ void rope_bf16_inplace(unsigned short* __restrict__ X,
                                  int rows, int nheads) {
    long long total = (long long)rows * nheads * 64;
    long long idx = (long long)blockIdx.x * blockDim.x + threadIdx.x;
    if (idx >= total) return;
    int i = (int)(idx % 64);
    int h = (int)((idx / 64) % nheads);
    long long row = idx / ((long long)64 * nheads);
    int s = (int)(row % 1024);

    float invf = powf(10000.0f, -((float)i) / 64.0f);
    float ang = (float)s * invf;
    float c = cosf(ang), sn = sinf(ang);

    long long off = row * (long long)nheads * 128 + (long long)h * 128;
    float q1 = bf2f(X[off + i]);
    float q2 = bf2f(X[off + i + 64]);
    X[off + i]      = f2bf(q1 * c - q2 * sn);
    X[off + i + 64] = f2bf(q2 * c + q1 * sn);
}

// fp32 in -> bf16 out RoPE (used for K after fake-quant)
__global__ void rope_f32_to_bf16(const float* __restrict__ X,
                                 unsigned short* __restrict__ Y,
                                 int rows, int nheads) {
    long long total = (long long)rows * nheads * 64;
    long long idx = (long long)blockIdx.x * blockDim.x + threadIdx.x;
    if (idx >= total) return;
    int i = (int)(idx % 64);
    int h = (int)((idx / 64) % nheads);
    long long row = idx / ((long long)64 * nheads);
    int s = (int)(row % 1024);

    float invf = powf(10000.0f, -((float)i) / 64.0f);
    float ang = (float)s * invf;
    float c = cosf(ang), sn = sinf(ang);

    long long off = row * (long long)nheads * 128 + (long long)h * 128;
    float q1 = X[off + i];
    float q2 = X[off + i + 64];
    Y[off + i]      = f2bf(q1 * c - q2 * sn);
    Y[off + i + 64] = f2bf(q2 * c + q1 * sn);
}

// ---------------- V transpose: (B*S, 8*128) f32 -> (B, kvh, 128, 1024) bf16 ----

__global__ void vtrans(const float* __restrict__ V, unsigned short* __restrict__ Vt) {
    long long idx = (long long)blockIdx.x * blockDim.x + threadIdx.x;
    if (idx >= (long long)2048 * 1024) return;
    int row = (int)(idx >> 10);       // b*1024 + s
    int col = (int)(idx & 1023);      // kvh*128 + d
    int b = row >> 10, s = row & 1023;
    int kh = col >> 7, d = col & 127;
    Vt[(((long long)b * 8 + kh) * 128 + d) * 1024 + s] = f2bf(V[idx]);
}

// ---------------- attention: one wave per (b, h, 16-row q block) ----------------
// dyn LDS: float sc[16][1024]  (64 KB). Output written as bf16.

__global__ __launch_bounds__(32) void attn16(const unsigned short* __restrict__ Qb,
                                             const unsigned short* __restrict__ Kb,
                                             const unsigned short* __restrict__ Vt,
                                             unsigned short* __restrict__ O) {
    extern __shared__ float sc[];
    const int lane = threadIdx.x;
    const int hf = lane >> 4;
    const int lr = lane & 15;

    const int qb = blockIdx.x & 63;
    const int h  = (blockIdx.x >> 6) & 31;
    const int b  = blockIdx.x >> 11;
    const int kvh = h >> 2;
    const int q0 = qb * 16;
    const int L  = q0 + 16;
    const int Lp = (L + 31) & ~31;
    const float scale = 0.08838834764831845f;   // 1/sqrt(128)

    const unsigned short* qrow = Qb + ((long long)(b * 1024 + q0 + lr)) * 4096 + h * 128;
    v16bf qa[4];
#pragma unroll
    for (int dc = 0; dc < 4; ++dc) qa[dc] = frag_from_bf16(qrow + dc * 32, hf);

    // pass 1: scores -> LDS with scale + causal mask
    for (int k0 = 0; k0 < L; k0 += 16) {
        const unsigned short* krow =
            Kb + ((long long)(b * 1024 + k0 + lr)) * 1024 + kvh * 128;
        v8f c = {};
#pragma unroll
        for (int dc = 0; dc < 4; ++dc)
            c = wmma_bf16(qa[dc], frag_from_bf16(krow + dc * 32, hf), c);
#pragma unroll
        for (int v = 0; v < 8; ++v) {
            int row = v + hf * 8;
            int key = k0 + lr;
            float val = (key <= q0 + row) ? c[v] * scale : -3.4028234663852886e38f;
            sc[row * 1024 + key] = val;
        }
    }
    if (Lp != L) {
        for (int t = lane; t < 256; t += 32)
            sc[(t >> 4) * 1024 + L + (t & 15)] = -3.4028234663852886e38f;
    }
    __syncthreads();

    // pass 2: exact softmax per row (2 lanes per row)
    float m = -3.4028234663852886e38f;
    for (int k = hf; k < L; k += 2) m = fmaxf(m, sc[lr * 1024 + k]);
    m = fmaxf(m, __shfl_xor(m, 16));
    float sum = 0.0f;
    for (int k = hf; k < L; k += 2) sum += expf(sc[lr * 1024 + k] - m);
    sum += __shfl_xor(sum, 16);
    float inv = 1.0f / sum;
    for (int k = hf; k < Lp; k += 2) sc[lr * 1024 + k] = expf(sc[lr * 1024 + k] - m) * inv;
    __syncthreads();

    // pass 3: O = P x V
    v8f o[8] = {};
    for (int k0 = 0; k0 < Lp; k0 += 32) {
        v16bf pa = frag_from_f32(&sc[lr * 1024 + k0], hf);
#pragma unroll
        for (int dt = 0; dt < 8; ++dt) {
            const unsigned short* vrow =
                Vt + (((long long)b * 8 + kvh) * 128 + dt * 16 + lr) * 1024 + k0;
            o[dt] = wmma_bf16(pa, frag_from_bf16(vrow, hf), o[dt]);
        }
    }
#pragma unroll
    for (int dt = 0; dt < 8; ++dt)
#pragma unroll
        for (int v = 0; v < 8; ++v) {
            int row = v + hf * 8;
            int d = dt * 16 + lr;
            O[((long long)(b * 1024 + q0 + row)) * 4096 + h * 128 + d] = f2bf(o[dt][v]);
        }
}

// ---------------- launch ----------------

extern "C" void kernel_launch(void* const* d_in, const int* in_sizes, int n_in,
                              void* d_out, int out_size, void* d_ws, size_t ws_size,
                              hipStream_t stream) {
    const float* hidden = (const float*)d_in[0];
    const float* Wq = (const float*)d_in[1];
    const float* Wk = (const float*)d_in[2];
    const float* Wv = (const float*)d_in[3];
    const float* Wo = (const float*)d_in[4];
    float* out = (float*)d_out;

    const size_t MiB = 1024 * 1024;
    char* ws = (char*)d_ws;
    unsigned short* Xb  = (unsigned short*)(ws);               //  16 MiB  hidden bf16
    unsigned short* Wqb = (unsigned short*)(ws + 16  * MiB);   //  32 MiB
    unsigned short* Wkb = (unsigned short*)(ws + 48  * MiB);   //   8 MiB
    unsigned short* Wvb = (unsigned short*)(ws + 56  * MiB);   //   8 MiB
    unsigned short* Wob = (unsigned short*)(ws + 64  * MiB);   //  32 MiB
    unsigned short* Qb  = (unsigned short*)(ws + 96  * MiB);   //  16 MiB  Q bf16 (RoPE in place)
    float*          Kf  = (float*)         (ws + 112 * MiB);   //   8 MiB  K f32
    float*          Vf  = (float*)         (ws + 120 * MiB);   //   8 MiB  V f32
    unsigned short* Kbq = (unsigned short*)(ws + 128 * MiB);   //   4 MiB  K bf16 (post RoPE)
    unsigned short* Vt  = (unsigned short*)(ws + 132 * MiB);   //   4 MiB  V^T bf16
    unsigned short* Ab  = (unsigned short*)(ws + 136 * MiB);   //  16 MiB  attn bf16

    // bulk fp32 -> bf16 converts (each element converted exactly once)
    cvt_f32_bf16<<<  8192, 256, 0, stream>>>(hidden, Xb,  (long long)2048 * 4096 / 4);
    cvt_f32_bf16<<< 16384, 256, 0, stream>>>(Wq,     Wqb, (long long)4096 * 4096 / 4);
    cvt_f32_bf16<<<  4096, 256, 0, stream>>>(Wk,     Wkb, (long long)1024 * 4096 / 4);
    cvt_f32_bf16<<<  4096, 256, 0, stream>>>(Wv,     Wvb, (long long)1024 * 4096 / 4);
    cvt_f32_bf16<<< 16384, 256, 0, stream>>>(Wo,     Wob, (long long)4096 * 4096 / 4);

    // projections (pure bf16 WMMA inner loops)
    gemm_bf16<true ><<<dim3(64, 32), 32, 0, stream>>>(Xb, Wqb, Qb, 2048, 4096, 4096);
    gemm_bf16<false><<<dim3(16, 32), 32, 0, stream>>>(Xb, Wkb, Kf, 2048, 1024, 4096);
    gemm_bf16<false><<<dim3(16, 32), 32, 0, stream>>>(Xb, Wvb, Vf, 2048, 1024, 4096);

    // fake quant: K along axis 0 (per column, n=2048), V along axis -1 (per row, n=1024)
    fake_quant<<<1024, 256, 0, stream>>>(Kf, 2048, 1LL, 1024LL);
    fake_quant<<<2048, 256, 0, stream>>>(Vf, 1024, 1024LL, 1LL);

    // RoPE + layout conversions
    rope_bf16_inplace<<<16384, 256, 0, stream>>>(Qb, 2048, 32);
    rope_f32_to_bf16 <<< 4096, 256, 0, stream>>>(Kf, Kbq, 2048, 8);
    vtrans<<<8192, 256, 0, stream>>>(Vf, Vt);

    // attention (dyn LDS 64KB per 1-wave workgroup)
    attn16<<<4096, 32, 65536, stream>>>(Qb, Kbq, Vt, Ab);

    // output projection
    gemm_bf16<false><<<dim3(64, 32), 32, 0, stream>>>(Ab, Wob, out, 2048, 4096, 4096);
}
